// DecoupledContrastiveLoss_9680856285462
// MI455X (gfx1250) — matrix operand order repaired and checked by reading, not compiled, not run on hardware
//
#include <hip/hip_runtime.h>
#include <stdint.h>

// ---------------------------------------------------------------------------
// DecoupledContrastiveLoss for MI455X (gfx1250, wave32, WMMA).
//
// B=8192 rows, D=512 features. Four fused 8192x8192x512 GEMM-reduce passes
// using v_wmma_f32_16x16x32_f16; sim tiles never leave registers (saves ~2 GB
// of HBM round-trip vs. materializing the four 8192^2 sim matrices).
//
// Parallelism: column dimension split into SEG=4 segments -> 512 workgroups
// (2048 waves) per GEMM pass instead of 128, so the WGP array is filled and
// load-wait stalls are hidden by co-resident waves. Per-segment partial row
// statistics are merged deterministically in the finalize kernel (no float
// atomics -> bitwise identical across graph replays).
// ---------------------------------------------------------------------------

#define Bn 8192
#define Dk 512
#define SEG 4                       // column segments per GEMM pass
#define CT_PER_SEG (Bn / 16 / SEG)  // 128 column tiles per segment
#define INV_T (1.0f / 0.07f)

typedef __attribute__((ext_vector_type(16))) _Float16 v16h;
typedef __attribute__((ext_vector_type(8)))  _Float16 v8h;
typedef __attribute__((ext_vector_type(8)))  float    v8f;

union HF16 { v16h v; v8h h[2]; };

// A-matrix fragment (16x32 f16, row-major source, ld=Dk).
// Lane L: row = L&15; kb = (L<16)?0:8. VGPRs 0-3 hold K=kb..kb+7,
// VGPRs 4-7 hold K=kb+16..kb+23 -> two contiguous 16-byte loads.
__device__ __forceinline__ v16h load_a_frag(const _Float16* __restrict__ rowptr,
                                            int c, int kb) {
  HF16 f;
  f.h[0] = *(const v8h*)(rowptr + c * 32 + kb);
  f.h[1] = *(const v8h*)(rowptr + c * 32 + kb + 16);
  return f.v;
}

// B-matrix fragment (32x16 f16) from a column-major (transposed) staging
// buffer xT[Dk][Bn]. Lane L holds N = cbase+(L&15); kb = (L<16)?0:8;
// elements 0-7 hold K=c*32+kb..kb+7, elements 8-15 hold K=c*32+kb+16..kb+23.
__device__ __forceinline__ v16h load_b_frag(const _Float16* __restrict__ xT,
                                            int c, int lane, int cbase) {
  const _Float16* p = xT + (size_t)(c * 32 + ((lane >> 4) * 8)) * Bn + cbase + (lane & 15);
  v16h f;
#pragma unroll
  for (int s = 0; s < 8; ++s) {
    f[s] = p[(size_t)s * Bn];
    f[8 + s] = p[(size_t)(16 + s) * Bn];
  }
  return f;
}

// ---------------------------------------------------------------------------
// Prep: per-row inverse L2 norm + f16 staging (row-major and transposed).
// One wave32 per row; 8 rows per 256-thread block.
// ---------------------------------------------------------------------------
__global__ __launch_bounds__(256) void prep_kernel(const float* __restrict__ x,
                                                   _Float16* __restrict__ xh,
                                                   _Float16* __restrict__ xhT,
                                                   float* __restrict__ invn) {
  const int lane = threadIdx.x & 31;
  const int row  = blockIdx.x * 8 + (threadIdx.x >> 5);
  const float* p = x + (size_t)row * Dk;
  float xv[16];
  float ss = 0.f;
#pragma unroll
  for (int t = 0; t < 16; ++t) {
    xv[t] = p[t * 32 + lane];
    ss += xv[t] * xv[t];
  }
#pragma unroll
  for (int m = 1; m <= 16; m <<= 1) ss += __shfl_xor(ss, m);
  const float r = rsqrtf(ss);
  if (lane == 0) invn[row] = r;
#pragma unroll
  for (int t = 0; t < 16; ++t) {
    const int c = t * 32 + lane;
    const _Float16 h = (_Float16)xv[t];
    xh[(size_t)row * Dk + c] = h;
    xhT[(size_t)c * Bn + row] = h;  // transposed copy for the B operand
  }
}

// ---------------------------------------------------------------------------
// Fused sim = (A @ B^T)/T with masked-LSE row statistics over one column
// segment. Grid: (Bn/64, SEG); 128 threads (4 waves) per block. Each wave owns
// a 16-row strip, keeps all K=512 A fragments in VGPRs, sweeps 128 column
// tiles of its segment. Accumulator layout (32-bit C/D 16x16): half hh=L>=16;
// element g maps to row i = rbase+g+8*hh, col j = cbase+(L&15).
// Outputs are per-segment partials at [seg*Bn + i].
// ---------------------------------------------------------------------------
__global__ __launch_bounds__(128) void cross_fused(
    const _Float16* __restrict__ ah, const _Float16* __restrict__ bT,
    const float* __restrict__ inva, const float* __restrict__ invb,
    const int* __restrict__ mid,
    float* __restrict__ rowtot, float* __restrict__ rowpos,
    int* __restrict__ rowcnt, int* __restrict__ rowpred,
    float* __restrict__ rowbv, int* __restrict__ rowfm) {
  const int lane  = threadIdx.x & 31;
  const int wave  = threadIdx.x >> 5;
  const int rbase = blockIdx.x * 64 + wave * 16;
  const int seg   = blockIdx.y;
  const int ct0   = seg * CT_PER_SEG;
  const int hh    = lane >> 4;
  const int kb    = hh ? 8 : 0;
  const int ln    = lane & 15;

  const _Float16* arow = ah + (size_t)(rbase + ln) * Dk;
  v16h a[16];
#pragma unroll
  for (int c = 0; c < 16; ++c) a[c] = load_a_frag(arow, c, kb);

  float inva_r[8];
  int   mid_r[8];
#pragma unroll
  for (int g = 0; g < 8; ++g) {
    const int i = rbase + g + hh * 8;
    inva_r[g] = inva[i];
    mid_r[g]  = mid[i];
  }

  float rt[8], rp[8], rbv[8];
  int   rc[8], rfm[8], rbj[8];
#pragma unroll
  for (int g = 0; g < 8; ++g) {
    rt[g] = 0.f; rp[g] = 0.f; rbv[g] = -3.4e38f;
    rc[g] = 0;   rfm[g] = 0x7fffffff; rbj[g] = 0;
  }

  for (int ct = ct0; ct < ct0 + CT_PER_SEG; ++ct) {
    const int cbase = ct * 16;
    if (ct + 1 < ct0 + CT_PER_SEG)  // pull next tile's B stripe toward caches
      __builtin_prefetch(bT + (size_t)lane * Bn + cbase + 16, 0, 1);

    v8f acc = {};
#pragma unroll
    for (int c = 0; c < 16; ++c) {
      const v16h b = load_b_frag(bT, c, lane, cbase);
      acc = __builtin_amdgcn_wmma_f32_16x16x32_f16(
          /*neg_a=*/false, a[c], /*neg_b=*/false, b,
          /*c_mod=*/(short)0, acc, /*reuse_a=*/false, /*reuse_b=*/false);
    }

    const int   j    = cbase + ln;
    const float sj   = invb[j] * INV_T;
    const int   midj = mid[j];
#pragma unroll
    for (int g = 0; g < 8; ++g) {
      const float s = acc[g] * inva_r[g] * sj;  // normalized sim / T
      const float e = __expf(s);
      rt[g] += e;
      if (mid_r[g] == midj) {
        rp[g] += e;
        rc[g] += 1;
        if (j < rfm[g]) rfm[g] = j;  // first positive (== argmax of mask row)
      }
      if (s > rbv[g]) { rbv[g] = s; rbj[g] = j; }  // first-max semantics
    }
  }

  // Reduce across the 16 lanes of each half (row i lives in one half).
#pragma unroll
  for (int g = 0; g < 8; ++g) {
#pragma unroll
    for (int m = 1; m <= 8; m <<= 1) {
      rt[g] += __shfl_xor(rt[g], m);
      rp[g] += __shfl_xor(rp[g], m);
      rc[g] += __shfl_xor(rc[g], m);
      const int ofm = __shfl_xor(rfm[g], m);
      if (ofm < rfm[g]) rfm[g] = ofm;
      const float obv = __shfl_xor(rbv[g], m);
      const int   obj = __shfl_xor(rbj[g], m);
      if (obv > rbv[g] || (obv == rbv[g] && obj < rbj[g])) {
        rbv[g] = obv; rbj[g] = obj;
      }
    }
    if (ln == 0) {
      const size_t i = (size_t)seg * Bn + rbase + g + hh * 8;
      rowtot[i]  = rt[g];
      rowpos[i]  = rp[g];
      rowcnt[i]  = rc[g];
      rowpred[i] = rbj[g];
      rowbv[i]   = rbv[g];
      rowfm[i]   = rfm[g];
    }
  }
}

// ---------------------------------------------------------------------------
// Fused instance sim = (X @ X^T)/T: per-segment row exp-sums + diagonal.
// ---------------------------------------------------------------------------
__global__ __launch_bounds__(128) void inst_fused(
    const _Float16* __restrict__ xh, const _Float16* __restrict__ xhT,
    const float* __restrict__ invn,
    float* __restrict__ rowtot, float* __restrict__ diag) {
  const int lane  = threadIdx.x & 31;
  const int wave  = threadIdx.x >> 5;
  const int rbase = blockIdx.x * 64 + wave * 16;
  const int seg   = blockIdx.y;
  const int ct0   = seg * CT_PER_SEG;
  const int hh    = lane >> 4;
  const int kb    = hh ? 8 : 0;
  const int ln    = lane & 15;

  const _Float16* arow = xh + (size_t)(rbase + ln) * Dk;
  v16h a[16];
#pragma unroll
  for (int c = 0; c < 16; ++c) a[c] = load_a_frag(arow, c, kb);

  float invn_r[8];
#pragma unroll
  for (int g = 0; g < 8; ++g) invn_r[g] = invn[rbase + g + hh * 8];

  float rt[8], dg[8];
#pragma unroll
  for (int g = 0; g < 8; ++g) { rt[g] = 0.f; dg[g] = 0.f; }

  for (int ct = ct0; ct < ct0 + CT_PER_SEG; ++ct) {
    const int cbase = ct * 16;
    if (ct + 1 < ct0 + CT_PER_SEG)
      __builtin_prefetch(xhT + (size_t)lane * Bn + cbase + 16, 0, 1);

    v8f acc = {};
#pragma unroll
    for (int c = 0; c < 16; ++c) {
      const v16h b = load_b_frag(xhT, c, lane, cbase);
      acc = __builtin_amdgcn_wmma_f32_16x16x32_f16(
          false, a[c], false, b, (short)0, acc, false, false);
    }
    const int   j  = cbase + ln;
    const float sj = invn[j] * INV_T;
#pragma unroll
    for (int g = 0; g < 8; ++g) {
      const int   i = rbase + g + hh * 8;
      const float s = acc[g] * invn_r[g] * sj;
      rt[g] += __expf(s);
      if (j == i) dg[g] = s;  // diagonal: log(exp(s_ii)) = s_ii
    }
  }

#pragma unroll
  for (int g = 0; g < 8; ++g) {
#pragma unroll
    for (int m = 1; m <= 8; m <<= 1) {
      rt[g] += __shfl_xor(rt[g], m);
      dg[g] += __shfl_xor(dg[g], m);  // at most one lane holds the diagonal
    }
    if (ln == 0) {
      const size_t i = (size_t)seg * Bn + rbase + g + hh * 8;
      rowtot[i] = rt[g];
      diag[i]   = dg[g];
    }
  }
}

// ---------------------------------------------------------------------------
// Merge per-segment partials for one direction's row i.
// ---------------------------------------------------------------------------
struct RowStats { float tot, pos; int cnt, pred, fm; };

__device__ __forceinline__ RowStats merge_row(
    const float* tot, const float* pos, const int* cnt, const int* pred,
    const float* bv, const int* fm, int i) {
  RowStats r;
  r.tot = 0.f; r.pos = 0.f; r.cnt = 0; r.fm = 0x7fffffff;
  float best = -3.4e38f; r.pred = 0;
#pragma unroll
  for (int s = 0; s < SEG; ++s) {
    const size_t k = (size_t)s * Bn + i;
    r.tot += tot[k];
    r.pos += pos[k];
    r.cnt += cnt[k];
    if (fm[k] < r.fm) r.fm = fm[k];
    // segments are in increasing-j order; strict > keeps first-max semantics
    if (bv[k] > best) { best = bv[k]; r.pred = pred[k]; }
  }
  return r;
}

// ---------------------------------------------------------------------------
// Final scalar reduction -> 9 outputs.
// ---------------------------------------------------------------------------
__global__ __launch_bounds__(256) void finalize_kernel(
    const float* __restrict__ v2t_tot, const float* __restrict__ v2t_pos,
    const int* __restrict__ v2t_cnt,  const int* __restrict__ v2t_pred,
    const float* __restrict__ v2t_bv, const int* __restrict__ v2t_fm,
    const float* __restrict__ t2v_tot, const float* __restrict__ t2v_pos,
    const int* __restrict__ t2v_cnt,  const int* __restrict__ t2v_pred,
    const float* __restrict__ t2v_bv, const int* __restrict__ t2v_fm,
    const float* __restrict__ vv_tot, const float* __restrict__ vv_diag,
    const float* __restrict__ tt_tot, const float* __restrict__ tt_diag,
    float* __restrict__ out) {
  const int tid = threadIdx.x;
  float v2t = 0.f, t2v = 0.f, vi = 0.f, ti = 0.f;
  int npos = 0, a1 = 0, a2 = 0;
  for (int i = tid; i < Bn; i += 256) {
    const RowStats rv = merge_row(v2t_tot, v2t_pos, v2t_cnt, v2t_pred,
                                  v2t_bv, v2t_fm, i);
    const RowStats rt = merge_row(t2v_tot, t2v_pos, t2v_cnt, t2v_pred,
                                  t2v_bv, t2v_fm, i);
    npos += rv.cnt;  // num_pos = sum of the match matrix
    if (rv.cnt > 0 && rv.cnt < Bn) v2t += -(logf(rv.pos) - logf(rv.tot));
    if (rt.cnt > 0 && rt.cnt < Bn) t2v += -(logf(rt.pos) - logf(rt.tot));
    float vvt = 0.f, vvd = 0.f, ttt = 0.f, ttd = 0.f;
#pragma unroll
    for (int s = 0; s < SEG; ++s) {
      const size_t k = (size_t)s * Bn + i;
      vvt += vv_tot[k]; vvd += vv_diag[k];  // diag nonzero in one segment
      ttt += tt_tot[k]; ttd += tt_diag[k];
    }
    vi += -(vvd - logf(vvt));
    ti += -(ttd - logf(ttt));
    a1 += (rv.pred == rv.fm);
    a2 += (rt.pred == rt.fm);
  }
  __shared__ float sf[4][256];
  __shared__ int   si[3][256];
  sf[0][tid] = v2t; sf[1][tid] = t2v; sf[2][tid] = vi; sf[3][tid] = ti;
  si[0][tid] = npos; si[1][tid] = a1; si[2][tid] = a2;
  __syncthreads();
  for (int s = 128; s > 0; s >>= 1) {
    if (tid < s) {
#pragma unroll
      for (int q = 0; q < 4; ++q) sf[q][tid] += sf[q][tid + s];
#pragma unroll
      for (int q = 0; q < 3; ++q) si[q][tid] += si[q][tid + s];
    }
    __syncthreads();
  }
  if (tid == 0) {
    const float num_pos   = (float)si[0][0];
    const float v2t_loss  = sf[0][0] / num_pos;
    const float t2v_loss  = sf[1][0] / num_pos;
    const float cross     = 0.5f * (v2t_loss + t2v_loss);
    const float vi_loss   = sf[2][0] / (float)Bn;
    const float ti_loss   = sf[3][0] / (float)Bn;
    const float total     = cross + 0.5f * vi_loss + 0.5f * ti_loss;
    const float v2t_acc   = (float)si[1][0] / (float)Bn;
    const float t2v_acc   = (float)si[2][0] / (float)Bn;
    out[0] = total;    out[1] = cross;    out[2] = v2t_loss;
    out[3] = t2v_loss; out[4] = vi_loss;  out[5] = ti_loss;
    out[6] = v2t_acc;  out[7] = t2v_acc;  out[8] = 0.5f * (v2t_acc + t2v_acc);
  }
}

// ---------------------------------------------------------------------------
extern "C" void kernel_launch(void* const* d_in, const int* in_sizes, int n_in,
                              void* d_out, int out_size, void* d_ws,
                              size_t ws_size, hipStream_t stream) {
  (void)in_sizes; (void)n_in; (void)out_size; (void)ws_size;
  const float* vf  = (const float*)d_in[0];
  const float* tf  = (const float*)d_in[1];
  const int*   mid = (const int*)d_in[2];
  float* out = (float*)d_out;

  char* ws = (char*)d_ws;
  size_t off = 0;
  const size_t MAT  = (size_t)Bn * Dk * sizeof(_Float16);  // 8 MB each
  const size_t SB_F = (size_t)SEG * Bn * sizeof(float);    // segmented f32
  const size_t SB_I = (size_t)SEG * Bn * sizeof(int);      // segmented i32
  _Float16* vh  = (_Float16*)(ws + off); off += MAT;
  _Float16* th  = (_Float16*)(ws + off); off += MAT;
  _Float16* vhT = (_Float16*)(ws + off); off += MAT;
  _Float16* thT = (_Float16*)(ws + off); off += MAT;
  float* invv = (float*)(ws + off); off += Bn * sizeof(float);
  float* invt = (float*)(ws + off); off += Bn * sizeof(float);

  float* v2t_tot  = (float*)(ws + off); off += SB_F;
  float* v2t_pos  = (float*)(ws + off); off += SB_F;
  int*   v2t_cnt  = (int*)  (ws + off); off += SB_I;
  int*   v2t_pred = (int*)  (ws + off); off += SB_I;
  float* v2t_bv   = (float*)(ws + off); off += SB_F;
  int*   v2t_fm   = (int*)  (ws + off); off += SB_I;
  float* t2v_tot  = (float*)(ws + off); off += SB_F;
  float* t2v_pos  = (float*)(ws + off); off += SB_F;
  int*   t2v_cnt  = (int*)  (ws + off); off += SB_I;
  int*   t2v_pred = (int*)  (ws + off); off += SB_I;
  float* t2v_bv   = (float*)(ws + off); off += SB_F;
  int*   t2v_fm   = (int*)  (ws + off); off += SB_I;
  float* vv_tot   = (float*)(ws + off); off += SB_F;
  float* vv_diag  = (float*)(ws + off); off += SB_F;
  float* tt_tot   = (float*)(ws + off); off += SB_F;
  float* tt_diag  = (float*)(ws + off); off += SB_F;

  // 1) normalization factors + f16 staging (row-major and transposed)
  prep_kernel<<<Bn / 8, 256, 0, stream>>>(vf, vh, vhT, invv);
  prep_kernel<<<Bn / 8, 256, 0, stream>>>(tf, th, thT, invt);

  const dim3 gg(Bn / 64, SEG);
  // 2) fused GEMM + masked-LSE row stats: v2t = v@t^T, t2v = t@v^T
  cross_fused<<<gg, 128, 0, stream>>>(vh, thT, invv, invt, mid,
      v2t_tot, v2t_pos, v2t_cnt, v2t_pred, v2t_bv, v2t_fm);
  cross_fused<<<gg, 128, 0, stream>>>(th, vhT, invt, invv, mid,
      t2v_tot, t2v_pos, t2v_cnt, t2v_pred, t2v_bv, t2v_fm);

  // 3) fused instance terms: v@v^T, t@t^T
  inst_fused<<<gg, 128, 0, stream>>>(vh, vhT, invv, vv_tot, vv_diag);
  inst_fused<<<gg, 128, 0, stream>>>(th, thT, invt, tt_tot, tt_diag);

  // 4) merge segments + scalar reduction -> 9 outputs
  finalize_kernel<<<1, 256, 0, stream>>>(
      v2t_tot, v2t_pos, v2t_cnt, v2t_pred, v2t_bv, v2t_fm,
      t2v_tot, t2v_pos, t2v_cnt, t2v_pred, t2v_bv, t2v_fm,
      vv_tot, vv_diag, tt_tot, tt_diag, out);
}
